// TrimKVPhi3Attention_75179107549566
// MI455X (gfx1250) — matrix operand order repaired and checked by reading, not compiled, not run on hardware
//
#include <hip/hip_runtime.h>

typedef __attribute__((ext_vector_type(16))) __bf16 v16bf;
typedef __attribute__((ext_vector_type(8)))  float  v8f;

#define B_  2
#define SEQ 2048
#define HID 2048
#define NH_ 16
#define NKV_ 4
#define HD_ 128
#define RGI 128
#define QKVN 3072
#define SCALE_F 0.08838834764831845f   // 128^-0.5

// ---------------------------------------------------------------- conversions
__global__ void cvt_f32_bf16(const float* __restrict__ in, __bf16* __restrict__ out, size_t n) {
    size_t i = (size_t)blockIdx.x * blockDim.x + threadIdx.x;
    size_t stride = (size_t)gridDim.x * blockDim.x;
    for (; i < n; i += stride) out[i] = (__bf16)in[i];
}

// ---------------------------------------------------------------- GEMM C = A @ W^T
// A: [M,K] bf16 row-major, W: [N,K] bf16 row-major, C: [M,N] fp32.
// Block = 8 waves; wave w computes rows m0+w*16..+15, cols n0..n0+127.
__global__ __launch_bounds__(256)
void gemm_bf16_nt(const __bf16* __restrict__ A, const __bf16* __restrict__ W,
                  float* __restrict__ C, int M, int N, int K) {
    const int lane   = threadIdx.x & 31;
    const int wave   = threadIdx.x >> 5;
    const int lane16 = lane & 15;
    const int khalf  = lane >> 4;
    const int m0 = blockIdx.x * 128 + wave * 16;
    const int n0 = blockIdx.y * 128;

    v8f acc[8];
    #pragma unroll
    for (int j = 0; j < 8; ++j)
        #pragma unroll
        for (int r = 0; r < 8; ++r) acc[j][r] = 0.0f;

    const __bf16* arow = A + (size_t)(m0 + lane16) * K;
    for (int kt = 0; kt < K; kt += 32) {
        v16bf av;
        const __bf16* p0 = arow + kt + khalf * 16;
        #pragma unroll
        for (int i = 0; i < 8; ++i) { av[i] = p0[i]; av[8 + i] = p0[8 + i]; }
        #pragma unroll
        for (int j = 0; j < 8; ++j) {
            const __bf16* wp = W + (size_t)(n0 + j * 16 + lane16) * K + kt + khalf * 16;
            v16bf bv = *(const v16bf*)wp;
            acc[j] = __builtin_amdgcn_wmma_f32_16x16x32_bf16(
                false, av, false, bv, (short)0, acc[j], false, false);
        }
    }
    #pragma unroll
    for (int j = 0; j < 8; ++j) {
        int n = n0 + j * 16 + lane16;
        #pragma unroll
        for (int r = 0; r < 8; ++r) {
            int m = m0 + r + 8 * khalf;
            C[(size_t)m * N + n] = acc[j][r];
        }
    }
}

// ---------------------------------------------------------------- retention gate
// Ret[b,kv,s] = log_sigmoid( silu(hs @ w1^T + b1) @ w2^T + bias )
__global__ __launch_bounds__(128)
void gate_kernel(const float* __restrict__ hs, const float* __restrict__ w1,
                 const float* __restrict__ b1, const float* __restrict__ w2,
                 const float* __restrict__ bias, float* __restrict__ Ret) {
    __shared__ float hrow[HID];
    __shared__ float inter[RGI];
    const int t  = threadIdx.x;
    const int bs = blockIdx.x;           // b*SEQ + s
    const int b  = bs / SEQ, s = bs % SEQ;
    const float* hp = hs + (size_t)bs * HID;
    for (int i = t; i < HID; i += 128) hrow[i] = hp[i];
    __syncthreads();
    float acc = b1[t];
    const float* wr = w1 + (size_t)t * HID;
    for (int k = 0; k < HID; k += 4)
        acc += hrow[k] * wr[k] + hrow[k+1] * wr[k+1] + hrow[k+2] * wr[k+2] + hrow[k+3] * wr[k+3];
    inter[t] = acc / (1.0f + __expf(-acc));     // silu
    __syncthreads();
    if (t < NKV_) {
        float g = bias[t];
        const float* w2r = w2 + t * RGI;
        for (int k = 0; k < RGI; ++k) g += inter[k] * w2r[k];
        float ls = fminf(g, 0.0f) - log1pf(__expf(-fabsf(g)));   // log_sigmoid
        Ret[((size_t)b * NKV_ + t) * SEQ + s] = ls;
    }
}

// ---------------------------------------------------------------- inclusive cumsum over S (in place)
__global__ __launch_bounds__(1024)
void cumsum_kernel(float* __restrict__ L) {      // grid = B*NKV blocks
    __shared__ float part[1024];
    const int t = threadIdx.x;
    float* seq = L + (size_t)blockIdx.x * SEQ;
    float a0 = seq[2 * t], a1 = seq[2 * t + 1];
    float ps = a0 + a1;
    part[t] = ps;
    __syncthreads();
    for (int off = 1; off < 1024; off <<= 1) {
        float add = (t >= off) ? part[t - off] : 0.0f;
        __syncthreads();
        part[t] += add;
        __syncthreads();
    }
    float excl = part[t] - ps;
    seq[2 * t]     = excl + a0;
    seq[2 * t + 1] = excl + a0 + a1;
}

// ---------------------------------------------------------------- RoPE + repack to bf16
// Q -> [b,h,s,d], K -> [b,kv,s,d], V -> transposed [b,kv,d,s]
__global__ __launch_bounds__(128)
void rope_pack(const float* __restrict__ qkv, const float* __restrict__ cosb,
               const float* __restrict__ sinb, __bf16* __restrict__ Qb,
               __bf16* __restrict__ Kb, __bf16* __restrict__ Vtb) {
    const int d    = threadIdx.x;       // 0..127
    const int bs   = blockIdx.x;        // b*SEQ + s
    const int slot = blockIdx.y;        // 0..23 : 16 q heads, 4 k heads, 4 v heads
    const int b = bs / SEQ, s = bs % SEQ;
    const float* row = qkv + (size_t)bs * QKVN;
    if (slot < 16) {
        float c  = cosb[(size_t)bs * HD_ + d];
        float si = sinb[(size_t)bs * HD_ + d];
        const float* q = row + slot * HD_;
        float rot = (d < 64) ? -q[d + 64] : q[d - 64];
        Qb[(((size_t)(b * NH_ + slot)) * SEQ + s) * HD_ + d] = (__bf16)(q[d] * c + rot * si);
    } else if (slot < 20) {
        float c  = cosb[(size_t)bs * HD_ + d];
        float si = sinb[(size_t)bs * HD_ + d];
        int kv = slot - 16;
        const float* kk = row + NH_ * HD_ + kv * HD_;
        float rot = (d < 64) ? -kk[d + 64] : kk[d - 64];
        Kb[(((size_t)(b * NKV_ + kv)) * SEQ + s) * HD_ + d] = (__bf16)(kk[d] * c + rot * si);
    } else {
        int kv = slot - 20;
        float x = row[(NH_ + NKV_) * HD_ + kv * HD_ + d];
        Vtb[(((size_t)(b * NKV_ + kv)) * HD_ + d) * SEQ + s] = (__bf16)x;
    }
}

// ---------------------------------------------------------------- flash attention with retention decay
// One wave per (b, h, 16 query rows). Out -> bf16 [b, s, h*HD + d].
__global__ __launch_bounds__(32)
void attn_kernel(const __bf16* __restrict__ Q, const __bf16* __restrict__ Km,
                 const __bf16* __restrict__ Vt, const float* __restrict__ L,
                 __bf16* __restrict__ Obf) {
    const int lane   = threadIdx.x;
    const int lane16 = lane & 15;
    const int khalf  = lane >> 4;
    const int i0 = blockIdx.x * 16;
    const int h  = blockIdx.y;
    const int b  = blockIdx.z;
    const int kv = h >> 2;                      // G = NH/NKV = 4

    const __bf16* Qp = Q  + ((size_t)(b * NH_  + h ) * SEQ) * HD_;
    const __bf16* Kp = Km + ((size_t)(b * NKV_ + kv) * SEQ) * HD_;
    const __bf16* Vp = Vt + ((size_t)(b * NKV_ + kv) * HD_) * SEQ;
    const float*  Lp = L  +  (size_t)(b * NKV_ + kv) * SEQ;

    // Q fragments (A layout), resident for the whole tile
    v16bf qa[4];
    {
        const __bf16* qrow = Qp + (size_t)(i0 + lane16) * HD_;
        #pragma unroll
        for (int kt = 0; kt < 4; ++kt) {
            const __bf16* p0 = qrow + kt * 32 + khalf * 16;
            #pragma unroll
            for (int i = 0; i < 8; ++i) { qa[kt][i] = p0[i]; qa[kt][8 + i] = p0[8 + i]; }
        }
    }
    float Lrow[8];
    #pragma unroll
    for (int r = 0; r < 8; ++r) Lrow[r] = Lp[i0 + r + 8 * khalf];

    v8f o[8];
    #pragma unroll
    for (int t = 0; t < 8; ++t)
        #pragma unroll
        for (int r = 0; r < 8; ++r) o[t][r] = 0.0f;
    float mrow[8], lrow[8];
    #pragma unroll
    for (int r = 0; r < 8; ++r) { mrow[r] = -3.0e38f; lrow[r] = 0.0f; }

    __shared__ __bf16 pT[16 * 32];

    const int jend = i0 + 15;
    for (int j0 = 0; j0 <= jend; j0 += 32) {
        // S = Q K^T over HD=128 (two 16-wide key tiles)
        v8f s0, s1;
        #pragma unroll
        for (int r = 0; r < 8; ++r) { s0[r] = 0.0f; s1[r] = 0.0f; }
        const __bf16* k0row = Kp + (size_t)(j0 + lane16) * HD_;
        const __bf16* k1row = k0row + (size_t)16 * HD_;
        #pragma unroll
        for (int kt = 0; kt < 4; ++kt) {
            v16bf b0 = *(const v16bf*)(k0row + kt * 32 + khalf * 16);
            v16bf b1 = *(const v16bf*)(k1row + kt * 32 + khalf * 16);
            s0 = __builtin_amdgcn_wmma_f32_16x16x32_bf16(false, qa[kt], false, b0, (short)0, s0, false, false);
            s1 = __builtin_amdgcn_wmma_f32_16x16x32_bf16(false, qa[kt], false, b1, (short)0, s1, false, false);
        }
        const float Lj0 = Lp[j0 + lane16];
        const float Lj1 = Lp[j0 + 16 + lane16];
        const int col0 = j0 + lane16, col1 = col0 + 16;
        float p0[8], p1[8], alpha[8];
        #pragma unroll
        for (int r = 0; r < 8; ++r) {
            int row = i0 + r + 8 * khalf;
            float v0 = s0[r] * SCALE_F + Lrow[r] - Lj0;
            float v1 = s1[r] * SCALE_F + Lrow[r] - Lj1;
            if (col0 > row) v0 = -3.0e38f;
            if (col1 > row) v1 = -3.0e38f;
            p0[r] = v0; p1[r] = v1;
        }
        // online softmax per row (rows live in 16-lane halves -> 16-wide reductions)
        #pragma unroll
        for (int r = 0; r < 8; ++r) {
            float mx = fmaxf(p0[r], p1[r]);
            for (int off = 1; off < 16; off <<= 1) mx = fmaxf(mx, __shfl_xor(mx, off, 32));
            float mn = fmaxf(mrow[r], mx);
            float a  = __expf(mrow[r] - mn);
            float e0 = __expf(p0[r] - mn);
            float e1 = __expf(p1[r] - mn);
            float rs = e0 + e1;
            for (int off = 1; off < 16; off <<= 1) rs += __shfl_xor(rs, off, 32);
            lrow[r] = lrow[r] * a + rs;
            mrow[r] = mn;
            alpha[r] = a;
            p0[r] = e0; p1[r] = e1;
        }
        #pragma unroll
        for (int t = 0; t < 8; ++t)
            #pragma unroll
            for (int r = 0; r < 8; ++r) o[t][r] *= alpha[r];
        // C-layout -> A-layout re-swizzle of P through LDS
        #pragma unroll
        for (int r = 0; r < 8; ++r) {
            int row = r + 8 * khalf;
            pT[row * 32 + lane16]      = (__bf16)p0[r];
            pT[row * 32 + 16 + lane16] = (__bf16)p1[r];
        }
        __syncthreads();
        v16bf pa;
        {
            const __bf16* pr = &pT[lane16 * 32 + khalf * 16];
            #pragma unroll
            for (int i = 0; i < 8; ++i) { pa[i] = pr[i]; pa[8 + i] = pr[8 + i]; }
        }
        __syncthreads();
        // O += P @ V   (V transposed: B fragment = 16 contiguous bf16 along s)
        const __bf16* vbase = Vp + (size_t)lane16 * SEQ + j0 + khalf * 16;
        #pragma unroll
        for (int t = 0; t < 8; ++t) {
            v16bf bv = *(const v16bf*)(vbase + (size_t)t * 16 * SEQ);
            o[t] = __builtin_amdgcn_wmma_f32_16x16x32_bf16(false, pa, false, bv, (short)0, o[t], false, false);
        }
    }
    // normalize + write [b, s, h*HD + d]
    #pragma unroll
    for (int r = 0; r < 8; ++r) {
        int row = i0 + r + 8 * khalf;
        float inv = 1.0f / lrow[r];
        size_t base = ((size_t)(b * SEQ + row)) * (NH_ * HD_) + (size_t)h * HD_;
        #pragma unroll
        for (int t = 0; t < 8; ++t)
            Obf[base + t * 16 + lane16] = (__bf16)(o[t][r] * inv);
    }
}

// ---------------------------------------------------------------- host launcher
extern "C" void kernel_launch(void* const* d_in, const int* in_sizes, int n_in,
                              void* d_out, int out_size, void* d_ws, size_t ws_size,
                              hipStream_t stream) {
    const float* hs     = (const float*)d_in[0];
    const float* cosb   = (const float*)d_in[1];
    const float* sinb   = (const float*)d_in[2];
    const float* qkv_w  = (const float*)d_in[3];
    const float* o_w    = (const float*)d_in[4];
    const float* rg_w1  = (const float*)d_in[5];
    const float* rg_b1  = (const float*)d_in[6];
    const float* rg_w2  = (const float*)d_in[7];
    const float* rg_bias= (const float*)d_in[8];
    float* out = (float*)d_out;

    const size_t BS = (size_t)B_ * SEQ;            // 4096
    char* ws = (char*)d_ws;
    size_t off = 0;
    auto take = [&](size_t bytes) -> void* {
        void* p = ws + off;
        off += (bytes + 255) & ~(size_t)255;
        return p;
    };
    __bf16* hs_bf   = (__bf16*)take(BS * HID * 2);
    __bf16* qkvw_bf = (__bf16*)take((size_t)QKVN * HID * 2);
    __bf16* ow_bf   = (__bf16*)take((size_t)HID * HID * 2);
    float*  Lbuf    = (float*) take((size_t)B_ * NKV_ * SEQ * 4);
    __bf16* q_bf    = (__bf16*)take((size_t)B_ * NH_  * SEQ * HD_ * 2);
    __bf16* k_bf    = (__bf16*)take((size_t)B_ * NKV_ * SEQ * HD_ * 2);
    __bf16* vt_bf   = (__bf16*)take((size_t)B_ * NKV_ * SEQ * HD_ * 2);
    float*  qkvf    = (float*) take(BS * QKVN * 4);
    __bf16* attn_bf = (__bf16*)qkvf;   // alias: qkvf dead after rope_pack
    (void)ws_size; (void)n_in; (void)in_sizes; (void)out_size;

    // 1) fp32 -> bf16
    cvt_f32_bf16<<<1024, 256, 0, stream>>>(hs,    hs_bf,   BS * HID);
    cvt_f32_bf16<<<1024, 256, 0, stream>>>(qkv_w, qkvw_bf, (size_t)QKVN * HID);
    cvt_f32_bf16<<<1024, 256, 0, stream>>>(o_w,   ow_bf,   (size_t)HID * HID);

    // 2) QKV projection: [4096,3072] = hs_bf @ qkv_w^T
    gemm_bf16_nt<<<dim3(BS / 128, QKVN / 128), 256, 0, stream>>>(
        hs_bf, qkvw_bf, qkvf, (int)BS, QKVN, HID);

    // 3) retention gate + cumsum
    gate_kernel<<<(int)BS, 128, 0, stream>>>(hs, rg_w1, rg_b1, rg_w2, rg_bias, Lbuf);
    cumsum_kernel<<<B_ * NKV_, 1024, 0, stream>>>(Lbuf);

    // 4) RoPE + repack (V transposed)
    rope_pack<<<dim3((int)BS, NH_ + 2 * NKV_), 128, 0, stream>>>(
        qkvf, cosb, sinb, q_bf, k_bf, vt_bf);

    // 5) flash attention with retention decay
    attn_kernel<<<dim3(SEQ / 16, NH_, B_), 32, 0, stream>>>(
        q_bf, k_bf, vt_bf, Lbuf, attn_bf);

    // 6) output projection: [4096,2048] = attn @ o_w^T
    gemm_bf16_nt<<<dim3(BS / 128, HID / 128), 256, 0, stream>>>(
        attn_bf, ow_bf, out, (int)BS, HID, HID);
}
